// LSTMUsingLSTMCell_35811437314457
// MI455X (gfx1250) — matrix operand order repaired and checked by reading, not compiled, not run on hardware
//
#include <hip/hip_runtime.h>
#include <hip/hip_bf16.h>
#include <math.h>

// ---------------------------------------------------------------------------
// 2-layer LSTM (B=128, T=80, UNITS=1024, EMB=100) for gfx1250 (MI455X).
// bf16 WMMA (v_wmma_f32_16x16x32_bf16) with fp32 accumulation.
// Persistent kernel: 64 WGs x 256 threads; each WG owns 16 hidden units
// (4 gate columns of 16); each wave owns one 16-row batch tile and keeps
// its cell state c in registers for all 80 timesteps. h ping-pongs through
// L2; layer-1 weights live in 256KB of LDS per WGP.
// ---------------------------------------------------------------------------

typedef __bf16 bf16;
typedef bf16  v8bf  __attribute__((ext_vector_type(8)));
typedef bf16  v16bf __attribute__((ext_vector_type(16)));
typedef float v8f   __attribute__((ext_vector_type(8)));

#define UNITS_  1024
#define NGATE_  4096
#define B_      128
#define T_      80
#define EMB_    100
#define EMBP_   128
#define NWG_    64
#define TPB_    256

// ---- workspace layout (bytes) ----
#define OFF_WT0  0u          // bf16 [4096][128]   W0^T padded
#define OFF_UT0  1048576u    // bf16 [4096][1024]  U0^T
#define OFF_WT1  9437184u    // bf16 [4096][1024]  W1^T
#define OFF_UT1  17825792u   // bf16 [4096][1024]  U1^T
#define OFF_X    26214400u   // bf16 [80][128][128] embedded+padded inputs
#define OFF_H0   28835840u   // bf16 [2][128][1024] h0 ping-pong
#define OFF_H1   29360128u   // bf16 [2][128][1024] h1 ping-pong
#define OFF_H1F  29884416u   // f32  [128][1024]    final h1
#define OFF_BAR  30408704u   // barrier counter

union V16 { v16bf v; v8bf h[2]; };

__device__ __forceinline__ float fsig(float x) { return 1.f / (1.f + __expf(-x)); }

// A-operand (16x32 bf16, row-major source [rows][ld]) per-lane load.
// lanes 0-15: elements 0..15 = K k0..k0+15 (contiguous)
// lanes 16-31: shifted by 16 in K.  Two global_load_b128 per lane.
__device__ __forceinline__ v16bf loadA(const bf16* __restrict__ base, int ld,
                                       int row, int k0, int hi8) {
  V16 a;
  const bf16* p = base + row * ld + k0 + hi8;
  a.h[0] = *(const v8bf*)(p);
  a.h[1] = *(const v8bf*)(p + 8);
  return a.v;
}

// B-operand (32x16 bf16) from transposed weights [N][K]: lane holds 16
// contiguous K values of one column.  Two b128 loads per lane.
__device__ __forceinline__ v16bf loadB(const bf16* __restrict__ base, int ld,
                                       int col, int k0, int hi16) {
  V16 b;
  const bf16* p = base + col * ld + k0 + hi16;
  b.h[0] = *(const v8bf*)(p);
  b.h[1] = *(const v8bf*)(p + 8);
  return b.v;
}

// Accumulate A[row-tile] x B over K into 4 gate accumulators.
__device__ __forceinline__ void gemm_acc(v8f acc[4],
                                         const bf16* __restrict__ A, int ldA,
                                         const bf16* __restrict__ Bw, int ldB,
                                         int colbase, int colstride,
                                         int K, int row, int hi8, int hi16) {
  for (int k0 = 0; k0 < K; k0 += 32) {
    v16bf a = loadA(A, ldA, row, k0, hi16);
#pragma unroll
    for (int g = 0; g < 4; ++g) {
      v16bf b = loadB(Bw, ldB, colbase + g * colstride, k0, hi16);
      acc[g] = __builtin_amdgcn_wmma_f32_16x16x32_bf16(
          false, a, false, b, (short)0, acc[g], false, false);
    }
  }
}

// Elementwise gates + cell/hidden update; c lives in registers across steps.
__device__ __forceinline__ void gates_update(v8f acc[4], v8f* c, float4 bias,
                                             bf16* __restrict__ hout,
                                             float* __restrict__ fout,
                                             int m0, int hi, int col) {
#pragma unroll
  for (int r = 0; r < 8; ++r) {
    float ig = fsig(acc[0][r] + bias.x);
    float fg = fsig(acc[1][r] + bias.y);
    float gg = tanhf(acc[2][r] + bias.z);
    float og = fsig(acc[3][r] + bias.w);
    float cn = fg * (*c)[r] + ig * gg;
    (*c)[r] = cn;
    float hn = og * tanhf(cn);
    int row = m0 + r + hi * 8;                 // C/D layout: lanes16-31 -> M+8
    hout[row * UNITS_ + col] = (bf16)hn;
    if (fout) fout[row * UNITS_ + col] = hn;
  }
}

// Monotonic-counter grid barrier (persistent kernel, all WGs resident).
__device__ __forceinline__ void gbar(unsigned* cnt, unsigned target) {
  __syncthreads();
  if (threadIdx.x == 0) {
    __threadfence();
    atomicAdd(cnt, 1u);
    while (__hip_atomic_load(cnt, __ATOMIC_ACQUIRE, __HIP_MEMORY_SCOPE_AGENT) < target)
      __builtin_amdgcn_s_sleep(2);
  }
  __syncthreads();
}

// ---------------------------------------------------------------------------
__global__ void k_init(unsigned* bar) {
  if (threadIdx.x == 0) bar[0] = 0u;
}

// Transpose + bf16-convert weights: [K][4096] f32 -> [4096][K] bf16.
__global__ void k_prep_w(const float* __restrict__ W0, const float* __restrict__ U0,
                         const float* __restrict__ W1, const float* __restrict__ U1,
                         bf16* __restrict__ WT0, bf16* __restrict__ UT0,
                         bf16* __restrict__ WT1, bf16* __restrict__ UT1) {
  int n = blockIdx.x;                          // 0..4095 gate column
  for (int k = threadIdx.x; k < UNITS_; k += blockDim.x) {
    UT0[n * UNITS_ + k] = (bf16)U0[k * NGATE_ + n];
    WT1[n * UNITS_ + k] = (bf16)W1[k * NGATE_ + n];
    UT1[n * UNITS_ + k] = (bf16)U1[k * NGATE_ + n];
    if (k < EMBP_)
      WT0[n * EMBP_ + k] = (k < EMB_) ? (bf16)W0[k * NGATE_ + n] : (bf16)0.f;
  }
}

// Embedding gather -> bf16, EMB padded 100 -> 128, layout [T][B][EMBP].
__global__ void k_prep_x(const int* __restrict__ tokens,
                         const float* __restrict__ emb, bf16* __restrict__ X) {
  int bid = blockIdx.x;                        // t*B + b
  int t = bid / B_, b = bid % B_;
  int e = threadIdx.x;                         // 0..127
  int tok = tokens[b * T_ + t];
  X[(t * B_ + b) * EMBP_ + e] = (e < EMB_) ? (bf16)emb[tok * EMB_ + e] : (bf16)0.f;
}

// ---------------------------------------------------------------------------
__global__ void __launch_bounds__(TPB_)
k_lstm(const bf16* __restrict__ WT0, const bf16* __restrict__ UT0,
       const bf16* __restrict__ WT1, const bf16* __restrict__ UT1,
       const bf16* __restrict__ X,
       const float* __restrict__ b0, const float* __restrict__ b1,
       bf16* __restrict__ h0buf, bf16* __restrict__ h1buf,
       float* __restrict__ h1f, unsigned* __restrict__ bar) {
  extern __shared__ bf16 lds[];                // 256KB: layer-1 weight slices
  bf16* sW1 = lds;                             // [64][1024]
  bf16* sU1 = lds + 64 * UNITS_;               // [64][1024]

  const int tid  = threadIdx.x;
  const int lane = tid & 31;
  const int wave = tid >> 5;                   // 0..7 -> batch row tile
  const int wg   = blockIdx.x;                 // 0..63 -> 16-unit slice
  const int un0  = wg * 16;
  const int m0   = wave * 16;
  const int lc   = lane & 15;
  const int hi   = lane >> 4;
  const int hi8  = hi * 8, hi16 = hi * 16;
  const int col  = un0 + lc;

  // Stage layer-1 weight slices (4 gates x 16 cols x 1024 K, bf16) into LDS.
  for (int i = tid; i < 64 * (UNITS_ / 8); i += TPB_) {
    int r = i >> 7;                            // 0..63 local row
    int ck = (i & 127) << 3;                   // K chunk of 8
    int grow = (r >> 4) * UNITS_ + un0 + (r & 15);
    *(v8bf*)&sW1[r * UNITS_ + ck] = *(const v8bf*)&WT1[(size_t)grow * UNITS_ + ck];
    *(v8bf*)&sU1[r * UNITS_ + ck] = *(const v8bf*)&UT1[(size_t)grow * UNITS_ + ck];
  }

  // Zero h ping buffers (parity 0) cooperatively.
  {
    unsigned* z0 = (unsigned*)h0buf;
    unsigned* z1 = (unsigned*)h1buf;
    for (int i = wg * TPB_ + tid; i < (B_ * UNITS_) / 2; i += NWG_ * TPB_) {
      z0[i] = 0u; z1[i] = 0u;
    }
  }
  unsigned tgt = NWG_;
  gbar(bar, tgt);

  // Per-lane gate biases (i, f, g, o columns of this lane's unit).
  float4 bias0 = make_float4(b0[0 * UNITS_ + col], b0[1 * UNITS_ + col],
                             b0[2 * UNITS_ + col], b0[3 * UNITS_ + col]);
  float4 bias1 = make_float4(b1[0 * UNITS_ + col], b1[1 * UNITS_ + col],
                             b1[2 * UNITS_ + col], b1[3 * UNITS_ + col]);

  // Persistent cell state in registers (16x16 tile each, fp32).
  v8f c0 = {0.f, 0.f, 0.f, 0.f, 0.f, 0.f, 0.f, 0.f};
  v8f c1 = {0.f, 0.f, 0.f, 0.f, 0.f, 0.f, 0.f, 0.f};

  for (int t = 0; t < T_; ++t) {
    const bf16* h0r = h0buf + (size_t)(t & 1) * (B_ * UNITS_);
    bf16*       h0w = h0buf + (size_t)((t + 1) & 1) * (B_ * UNITS_);
    const bf16* h1r = h1buf + (size_t)(t & 1) * (B_ * UNITS_);
    bf16*       h1w = h1buf + (size_t)((t + 1) & 1) * (B_ * UNITS_);
    const bf16* Xt  = X + (size_t)t * B_ * EMBP_;

    if (t + 1 < T_)  // warm L2/WGP$ for next step's inputs
      __builtin_prefetch(Xt + (size_t)B_ * EMBP_ + (size_t)(m0 + lc) * EMBP_, 0, 1);

    // ---- layer 0: z = X[t] @ W0 + h0 @ U0 + b0 ----
    v8f acc[4] = {c0 - c0, c0 - c0, c0 - c0, c0 - c0};
    acc[0] = acc[1] = acc[2] = acc[3] = v8f{0.f,0.f,0.f,0.f,0.f,0.f,0.f,0.f};
    gemm_acc(acc, Xt,  EMBP_,  WT0, EMBP_,  un0 + lc, UNITS_, EMBP_,  m0 + lc, hi8, hi16);
    gemm_acc(acc, h0r, UNITS_, UT0, UNITS_, un0 + lc, UNITS_, UNITS_, m0 + lc, hi8, hi16);
    gates_update(acc, &c0, bias0, h0w, nullptr, m0, hi, col);

    tgt += NWG_;
    gbar(bar, tgt);                            // h0(t) visible everywhere

    // ---- layer 1: z = h0(t) @ W1 + h1 @ U1 + b1  (B from LDS) ----
    acc[0] = acc[1] = acc[2] = acc[3] = v8f{0.f,0.f,0.f,0.f,0.f,0.f,0.f,0.f};
    gemm_acc(acc, h0w, UNITS_, sW1, UNITS_, lc, 16, UNITS_, m0 + lc, hi8, hi16);
    gemm_acc(acc, h1r, UNITS_, sU1, UNITS_, lc, 16, UNITS_, m0 + lc, hi8, hi16);
    gates_update(acc, &c1, bias1, h1w, (t == T_ - 1) ? h1f : nullptr, m0, hi, col);

    tgt += NWG_;
    gbar(bar, tgt);                            // h1(t) visible everywhere
  }
}

// Final head: out[b] = sigmoid(h1 . Wout + bout), 128 outputs.
__global__ void k_head(const float* __restrict__ h1f, const float* __restrict__ Wout,
                       const float* __restrict__ bout, float* __restrict__ out) {
  int b = threadIdx.x;
  if (b < B_) {
    float s = 0.f;
    for (int k = 0; k < UNITS_; ++k) s += h1f[b * UNITS_ + k] * Wout[k];
    out[b] = 1.f / (1.f + __expf(-(s + bout[0])));
  }
}

// ---------------------------------------------------------------------------
extern "C" void kernel_launch(void* const* d_in, const int* in_sizes, int n_in,
                              void* d_out, int out_size, void* d_ws, size_t ws_size,
                              hipStream_t stream) {
  const int*   tokens = (const int*)  d_in[0];
  const float* emb    = (const float*)d_in[1];
  const float* W0     = (const float*)d_in[2];
  const float* U0     = (const float*)d_in[3];
  const float* b0     = (const float*)d_in[4];
  const float* W1     = (const float*)d_in[5];
  const float* U1     = (const float*)d_in[6];
  const float* b1     = (const float*)d_in[7];
  const float* Wout   = (const float*)d_in[8];
  const float* bout   = (const float*)d_in[9];
  float* out = (float*)d_out;

  char* ws = (char*)d_ws;
  bf16*     WT0 = (bf16*)(ws + OFF_WT0);
  bf16*     UT0 = (bf16*)(ws + OFF_UT0);
  bf16*     WT1 = (bf16*)(ws + OFF_WT1);
  bf16*     UT1 = (bf16*)(ws + OFF_UT1);
  bf16*     X   = (bf16*)(ws + OFF_X);
  bf16*     h0b = (bf16*)(ws + OFF_H0);
  bf16*     h1b = (bf16*)(ws + OFF_H1);
  float*    h1f = (float*)(ws + OFF_H1F);
  unsigned* bar = (unsigned*)(ws + OFF_BAR);

  k_init  <<<1, 64, 0, stream>>>(bar);
  k_prep_w<<<NGATE_, 256, 0, stream>>>(W0, U0, W1, U1, WT0, UT0, WT1, UT1);
  k_prep_x<<<T_ * B_, EMBP_, 0, stream>>>(tokens, emb, X);
  k_lstm  <<<NWG_, TPB_, 2 * 64 * UNITS_ * (int)sizeof(bf16), stream>>>(
      WT0, UT0, WT1, UT1, X, b0, b1, h0b, h1b, h1f, bar);
  k_head  <<<1, 128, 0, stream>>>(h1f, Wout, bout, out);
}
